// MultiHeadAttention_1022202217286
// MI455X (gfx1250) — matrix operand order repaired and checked by reading, not compiled, not run on hardware
//
#include <hip/hip_runtime.h>

#define S_LEN   4096
#define D_MODEL 512
#define N_HEAD  8
#define D_HEAD  64
#define MQ      64      // query rows per attention workgroup
#define KSPLIT  4       // k-range split (partial-sum buffers, deterministic)

typedef __attribute__((ext_vector_type(16))) _Float16     v16h;
typedef __attribute__((ext_vector_type(8)))  _Float16     v8h;
typedef __attribute__((ext_vector_type(8)))  float        v8f;
typedef __attribute__((ext_vector_type(4)))  float        f32x4;
typedef __attribute__((ext_vector_type(4)))  unsigned int u32x4;
typedef __attribute__((ext_vector_type(2)))  unsigned int u32x2;

union AFrag  { v16h v; unsigned int u[8]; };
union BFrag  { v16h v; u32x4 q[2]; };
union PackH4 { _Float16 h[4]; u32x2 v; };
union H8     { v8h v; u32x4 q; };

// K-column for VGPR r of a 16x32 f16 A-fragment (contiguous order, matches B):
// lanes 0-15: V0..7 -> K={0,1},...,{14,15}
// lanes 16-31: +16
__device__ __forceinline__ int a_kcol(int r, int lane) {
  return 2 * r + ((lane >> 4) << 4);
}

__device__ __forceinline__ v8f wmma_f16(v16h a, v16h b, v8f c) {
  return __builtin_amdgcn_wmma_f32_16x16x32_f16(false, a, false, b,
                                                (short)0, c, false, false);
}

// ---------------------------------------------------------------------------
// QKV projection: Y = X(4096x512,f32) @ W^T(512x512,f32) -> f16, head-split.
// transposeV==0: out[h][s][d]   (Q, K)
// transposeV==1: out[h][d][s]   (V^T, so PV B-fragments are contiguous)
// ---------------------------------------------------------------------------
__global__ __launch_bounds__(256) void proj_qkv_kernel(
    const float* __restrict__ X, const float* __restrict__ W,
    _Float16* __restrict__ out, int transposeV) {
  __shared__ __align__(16) _Float16 Xs[128 * 32];
  __shared__ __align__(16) _Float16 Ws[128 * 32];
  const int tid  = threadIdx.x;
  const int lane = tid & 31;
  const int wid  = tid >> 5;
  const int m0 = blockIdx.x * 128;
  const int n0 = blockIdx.y * 128;
  const int wm = (wid >> 1) * 32;   // 4 row-groups
  const int wn = (wid & 1) * 64;    // 2 col-groups

  v8f acc[2][4] = {};

  for (int k0 = 0; k0 < D_MODEL; k0 += 32) {
    for (int i = 0; i < 4; ++i) {
      int c = tid + i * 256;             // 1024 float4 chunks
      int row = c >> 3;
      int cc  = (c & 7) * 4;
      f32x4 x = *(const f32x4*)&X[(size_t)(m0 + row) * D_MODEL + k0 + cc];
      PackH4 px;
      px.h[0] = (_Float16)x.x; px.h[1] = (_Float16)x.y;
      px.h[2] = (_Float16)x.z; px.h[3] = (_Float16)x.w;
      *(u32x2*)&Xs[row * 32 + cc] = px.v;
      f32x4 w = *(const f32x4*)&W[(size_t)(n0 + row) * D_MODEL + k0 + cc];
      PackH4 pw;
      pw.h[0] = (_Float16)w.x; pw.h[1] = (_Float16)w.y;
      pw.h[2] = (_Float16)w.z; pw.h[3] = (_Float16)w.w;
      *(u32x2*)&Ws[row * 32 + cc] = pw.v;
    }
    __syncthreads();

    AFrag a[2];
    for (int mt = 0; mt < 2; ++mt)
      for (int r = 0; r < 8; ++r)
        a[mt].u[r] = *(const unsigned int*)
            &Xs[(wm + mt * 16 + (lane & 15)) * 32 + a_kcol(r, lane)];

    for (int nt = 0; nt < 4; ++nt) {
      BFrag b;  // B[k][n] = W[n][k]: 16 contiguous k per lane
      const _Float16* src = &Ws[(wn + nt * 16 + (lane & 15)) * 32 + ((lane >> 4) * 16)];
      b.q[0] = *(const u32x4*)src;
      b.q[1] = *(const u32x4*)(src + 8);
      for (int mt = 0; mt < 2; ++mt)
        acc[mt][nt] = wmma_f16(a[mt].v, b.v, acc[mt][nt]);
    }
    __syncthreads();
  }

  for (int mt = 0; mt < 2; ++mt)
    for (int nt = 0; nt < 4; ++nt)
      for (int r = 0; r < 8; ++r) {
        int s = m0 + wm + mt * 16 + ((lane >> 4) * 8) + r;
        int n = n0 + wn + nt * 16 + (lane & 15);
        int h = n >> 6, d = n & 63;
        _Float16 val = (_Float16)acc[mt][nt][r];
        if (transposeV)
          out[((size_t)h * D_HEAD + d) * S_LEN + s] = val;
        else
          out[((size_t)h * S_LEN + s) * D_HEAD + d] = val;
      }
}

// ---------------------------------------------------------------------------
// Fused attention. Softmax is over the HEAD axis -> pointwise per (q,k):
// single pass over k, and the k-range splits freely (partial sums just add).
// Workgroup = 64 query rows x one k-slice; wave h = head h. K/V B-fragments
// are register-reused across 4 q-tiles (4x less L2 traffic than a 16-row tile).
// Each k-slice writes its own f16 partial buffer (deterministic, no atomics).
// ---------------------------------------------------------------------------
__global__ __launch_bounds__(256) void attn_kernel(
    const _Float16* __restrict__ Qf, const _Float16* __restrict__ Kf,
    const _Float16* __restrict__ Vt, _Float16* __restrict__ AO) {
  __shared__ __align__(16) float     scLDS[MQ * 32 * N_HEAD];  // [q*32+k][h] 64KB
  __shared__ __align__(16) _Float16  atLDS[N_HEAD * MQ * 32];  // [h][q][k]  32KB
  const int tid   = threadIdx.x;
  const int lane  = tid & 31;
  const int h     = tid >> 5;
  const int qbase = blockIdx.x * MQ;
  const int ks    = blockIdx.y;                      // k slice
  _Float16* AOp   = AO + (size_t)ks * S_LEN * D_MODEL;

  // Q fragments for 4 q-tiles x 2 d-halves; resident for the whole k loop.
  AFrag qa[4][2];
  for (int qt = 0; qt < 4; ++qt)
    for (int ds = 0; ds < 2; ++ds)
      for (int r = 0; r < 8; ++r) {
        int row  = qbase + qt * 16 + (lane & 15);
        int kcol = ds * 32 + a_kcol(r, lane);
        qa[qt][ds].u[r] = *(const unsigned int*)
            &Qf[((size_t)h * S_LEN + row) * D_HEAD + kcol];
      }

  v8f o[4][4] = {};
  const int qq    = tid >> 2;        // softmax: row 0..63
  const int kbase = (tid & 3) * 8;   // softmax: 8 consecutive k

  const int kb0 = ks * (S_LEN / 32 / KSPLIT);
  for (int kbl = 0; kbl < S_LEN / 32 / KSPLIT; ++kbl) {
    const int kb = kb0 + kbl;

    // ---- Phase A: S_h = Q_h @ K_h^T, 64x32 (q,k) tile; B reused over qt ----
    v8f sc[4][2] = {};
    for (int kt = 0; kt < 2; ++kt) {
      int kcol = kb * 32 + kt * 16 + (lane & 15);
      const _Float16* krow = &Kf[((size_t)h * S_LEN + kcol) * D_HEAD];
      for (int ds = 0; ds < 2; ++ds) {
        BFrag b;  // B[d][k] = K[k][d]: contiguous d per lane
        const _Float16* src = krow + ds * 32 + ((lane >> 4) * 16);
        b.q[0] = *(const u32x4*)src;
        b.q[1] = *(const u32x4*)(src + 8);
        for (int qt = 0; qt < 4; ++qt)
          sc[qt][kt] = wmma_f16(qa[qt][ds].v, b.v, sc[qt][kt]);
      }
    }
    if (kbl + 1 < S_LEN / 32 / KSPLIT) {
      __builtin_prefetch(
          &Kf[((size_t)h * S_LEN + (kb + 1) * 32 + (lane & 15)) * D_HEAD], 0, 1);
      __builtin_prefetch(
          &Vt[((size_t)h * D_HEAD + (lane & 15)) * S_LEN + (kb + 1) * 32], 0, 1);
    }
    for (int qt = 0; qt < 4; ++qt)
      for (int kt = 0; kt < 2; ++kt)
        for (int r = 0; r < 8; ++r) {
          int m = qt * 16 + ((lane >> 4) * 8) + r;
          int k = kt * 16 + (lane & 15);
          scLDS[(m * 32 + k) * 8 + h] = sc[qt][kt][r] * 0.125f;  // 1/sqrt(64)
        }
    __syncthreads();

    // ---- Phase B: softmax across 8 heads; 8 (q,k) positions per thread ----
    for (int g = 0; g < 2; ++g) {
      int k0 = kbase + g * 4;
      const f32x4* p = (const f32x4*)&scLDS[((size_t)qq * 32 + k0) * 8];
      float ex[4][8], inv[4];
      for (int pos = 0; pos < 4; ++pos) {
        f32x4 a = p[pos * 2], b = p[pos * 2 + 1];
        float vv[8] = { a.x, a.y, a.z, a.w, b.x, b.y, b.z, b.w };
        float mx = vv[0];
        for (int i = 1; i < 8; ++i) mx = fmaxf(mx, vv[i]);
        float sm = 0.f;
        for (int i = 0; i < 8; ++i) { ex[pos][i] = __expf(vv[i] - mx); sm += ex[pos][i]; }
        inv[pos] = 1.0f / sm;
      }
      for (int i = 0; i < 8; ++i) {
        PackH4 pk;
        pk.h[0] = (_Float16)(ex[0][i] * inv[0]);
        pk.h[1] = (_Float16)(ex[1][i] * inv[1]);
        pk.h[2] = (_Float16)(ex[2][i] * inv[2]);
        pk.h[3] = (_Float16)(ex[3][i] * inv[3]);
        *(u32x2*)&atLDS[i * (MQ * 32) + qq * 32 + k0] = pk.v;
      }
    }
    __syncthreads();

    // ---- Phase C: O_h += attn_h @ V_h; B(V) reused over 4 q-tiles ----
    AFrag af[4];
    for (int qt = 0; qt < 4; ++qt)
      for (int r = 0; r < 8; ++r)
        af[qt].u[r] = *(const unsigned int*)
            &atLDS[h * (MQ * 32) + (qt * 16 + (lane & 15)) * 32 + a_kcol(r, lane)];
    for (int nt = 0; nt < 4; ++nt) {
      BFrag b;  // B[k][d] = Vt[d][k]: contiguous k per lane
      int d = nt * 16 + (lane & 15);
      const _Float16* src =
          &Vt[((size_t)h * D_HEAD + d) * S_LEN + kb * 32 + ((lane >> 4) * 16)];
      b.q[0] = *(const u32x4*)src;
      b.q[1] = *(const u32x4*)(src + 8);
      for (int qt = 0; qt < 4; ++qt)
        o[qt][nt] = wmma_f16(af[qt].v, b.v, o[qt][nt]);
    }
    __syncthreads();  // protect scLDS/atLDS for next iteration
  }

  // Merged-head partial store: col = h*64 + d, f16.
  for (int qt = 0; qt < 4; ++qt)
    for (int nt = 0; nt < 4; ++nt)
      for (int r = 0; r < 8; ++r) {
        int s   = qbase + qt * 16 + ((lane >> 4) * 8) + r;
        int col = h * 64 + nt * 16 + (lane & 15);
        AOp[(size_t)s * D_MODEL + col] = (_Float16)o[qt][nt][r];
      }
}

// ---------------------------------------------------------------------------
// Output projection: d_out = (sum of KSPLIT f16 partials) @ Wo^T(f32) -> f32.
// Partials are summed during LDS staging with packed f16 vector adds.
// ---------------------------------------------------------------------------
__global__ __launch_bounds__(256) void proj_out_kernel(
    const _Float16* __restrict__ A, const float* __restrict__ W,
    float* __restrict__ out) {
  __shared__ __align__(16) _Float16 Xs[128 * 32];
  __shared__ __align__(16) _Float16 Ws[128 * 32];
  const int tid  = threadIdx.x;
  const int lane = tid & 31;
  const int wid  = tid >> 5;
  const int m0 = blockIdx.x * 128;
  const int n0 = blockIdx.y * 128;
  const int wm = (wid >> 1) * 32;
  const int wn = (wid & 1) * 64;
  const size_t SLICE = (size_t)S_LEN * D_MODEL;

  v8f acc[2][4] = {};

  for (int k0 = 0; k0 < D_MODEL; k0 += 32) {
    for (int i = 0; i < 2; ++i) {
      int c = tid + i * 256;             // 512 chunks of 8 f16 (16B)
      int row = c >> 2;
      int cc  = (c & 3) * 8;
      size_t off = (size_t)(m0 + row) * D_MODEL + k0 + cc;
      H8 a0, a1, a2, a3, s;
      a0.q = *(const u32x4*)&A[off];
      a1.q = *(const u32x4*)&A[off + SLICE];
      a2.q = *(const u32x4*)&A[off + 2 * SLICE];
      a3.q = *(const u32x4*)&A[off + 3 * SLICE];
      s.v = (a0.v + a1.v) + (a2.v + a3.v);   // v_pk_add_f16
      *(u32x4*)&Xs[row * 32 + cc] = s.q;
    }
    for (int i = 0; i < 4; ++i) {
      int c = tid + i * 256;
      int row = c >> 3;
      int cc  = (c & 7) * 4;
      f32x4 w = *(const f32x4*)&W[(size_t)(n0 + row) * D_MODEL + k0 + cc];
      PackH4 pw;
      pw.h[0] = (_Float16)w.x; pw.h[1] = (_Float16)w.y;
      pw.h[2] = (_Float16)w.z; pw.h[3] = (_Float16)w.w;
      *(u32x2*)&Ws[row * 32 + cc] = pw.v;
    }
    __syncthreads();

    AFrag a[2];
    for (int mt = 0; mt < 2; ++mt)
      for (int r = 0; r < 8; ++r)
        a[mt].u[r] = *(const unsigned int*)
            &Xs[(wm + mt * 16 + (lane & 15)) * 32 + a_kcol(r, lane)];

    for (int nt = 0; nt < 4; ++nt) {
      BFrag b;
      const _Float16* src = &Ws[(wn + nt * 16 + (lane & 15)) * 32 + ((lane >> 4) * 16)];
      b.q[0] = *(const u32x4*)src;
      b.q[1] = *(const u32x4*)(src + 8);
      for (int mt = 0; mt < 2; ++mt)
        acc[mt][nt] = wmma_f16(a[mt].v, b.v, acc[mt][nt]);
    }
    __syncthreads();
  }

  for (int mt = 0; mt < 2; ++mt)
    for (int nt = 0; nt < 4; ++nt)
      for (int r = 0; r < 8; ++r) {
        int s = m0 + wm + mt * 16 + ((lane >> 4) * 8) + r;
        int n = n0 + wn + nt * 16 + (lane & 15);
        out[(size_t)s * D_MODEL + n] = acc[mt][nt][r];
      }
}

// ---------------------------------------------------------------------------
extern "C" void kernel_launch(void* const* d_in, const int* in_sizes, int n_in,
                              void* d_out, int out_size, void* d_ws, size_t ws_size,
                              hipStream_t stream) {
  (void)in_sizes; (void)n_in; (void)out_size; (void)ws_size;
  const float* q  = (const float*)d_in[0];
  const float* k  = (const float*)d_in[1];
  const float* v  = (const float*)d_in[2];
  const float* Wq = (const float*)d_in[3];
  const float* Wk = (const float*)d_in[4];
  const float* Wv = (const float*)d_in[5];
  const float* Wo = (const float*)d_in[6];

  // Workspace: Qf16 | Kf16 | Vt_f16 (4MB each) | KSPLIT f16 partials (16MB)
  _Float16* Qf = (_Float16*)d_ws;
  _Float16* Kf = Qf + (size_t)N_HEAD * S_LEN * D_HEAD;
  _Float16* Vt = Kf + (size_t)N_HEAD * S_LEN * D_HEAD;
  _Float16* AO = Vt + (size_t)N_HEAD * S_LEN * D_HEAD;

  dim3 g(S_LEN / 128, D_MODEL / 128), b(256);
  proj_qkv_kernel<<<g, b, 0, stream>>>(q, Wq, Qf, 0);
  proj_qkv_kernel<<<g, b, 0, stream>>>(k, Wk, Kf, 0);
  proj_qkv_kernel<<<g, b, 0, stream>>>(v, Wv, Vt, 1);
  attn_kernel<<<dim3(S_LEN / MQ, KSPLIT), 256, 0, stream>>>(Qf, Kf, Vt, AO);
  proj_out_kernel<<<g, b, 0, stream>>>(AO, Wo, (float*)d_out);
}
